// MultiheadAttention_47751446397022
// MI455X (gfx1250) — matrix operand order repaired and checked by reading, not compiled, not run on hardware
//
#include <hip/hip_runtime.h>

// ---------------- types / helpers ----------------
typedef __attribute__((ext_vector_type(16))) __bf16        v16bf;
typedef __attribute__((ext_vector_type(8)))  float         v8f;
typedef __attribute__((ext_vector_type(4)))  unsigned int  u32x4;
typedef __attribute__((ext_vector_type(4)))  float         f32x4;

union Frag {                 // one 16x32 bf16 A-frag or 32x16 bf16 B-frag (per-lane view)
  v16bf          bf;
  u32x4          q[2];
  unsigned short h[16];
};

__device__ __forceinline__ unsigned short f32_to_bf16(float f) {
  unsigned int u = __builtin_bit_cast(unsigned int, f);
  u += 0x7FFFu + ((u >> 16) & 1u);           // round-to-nearest-even
  return (unsigned short)(u >> 16);
}
__device__ __forceinline__ unsigned int pack_bf16(float a, float b) {
  return (unsigned int)f32_to_bf16(a) | ((unsigned int)f32_to_bf16(b) << 16);
}

static constexpr int Bb = 2;
static constexpr int SS = 2048;
static constexpr int DD = 1024;
static constexpr int HH = 16;
static constexpr int HD = 64;
static constexpr int MM = Bb * SS;   // 4096
static constexpr int KK = DD;        // 1024
static constexpr int NN = DD;        // 1024

#define WMMA_BF16(A, B, C) \
  __builtin_amdgcn_wmma_f32_16x16x32_bf16(false, (A), false, (B), (short)0, (C), false, false)

// ---------------- kernel 0: f32 -> bf16 bulk converter (8 elems/thread) ----------------
__global__ __launch_bounds__(256) void cvt_bf16_kernel(
    const float* __restrict__ in, unsigned short* __restrict__ out) {
  const size_t i = ((size_t)blockIdx.x * blockDim.x + threadIdx.x) * 8;
  f32x4 x0 = *(const f32x4*)(in + i);
  f32x4 x1 = *(const f32x4*)(in + i + 4);
  u32x4 w;
  w.x = pack_bf16(x0.x, x0.y);
  w.y = pack_bf16(x0.z, x0.w);
  w.z = pack_bf16(x1.x, x1.y);
  w.w = pack_bf16(x1.z, x1.w);
  *(u32x4*)(out + i) = w;
}

// ---------------- unified GEMM: A(bf16)[M,K] @ W(bf16)[N,K]^T + bias ----------------
// one wave => 32x64 output tile (2 A-frags share each B-frag => 8 WMMA / 12 b128 loads)
// mode 0: bf16 row-major   mode 1: bf16 vT[b][h][d][s]   mode 2: f32 row-major
__global__ __launch_bounds__(256) void gemm_bf16_kernel(
    const unsigned short* __restrict__ X, const unsigned short* __restrict__ W,
    const float* __restrict__ bias, unsigned short* __restrict__ outb,
    float* __restrict__ outf, int mode) {
  const int lane  = threadIdx.x & 31;
  const int wave  = blockIdx.x * 8 + (threadIdx.x >> 5);
  const int mtile = wave >> 4;              // 128 m-tiles of 32 rows
  const int nblk  = wave & 15;              // 16 n-blocks of 64 cols
  const int half  = lane >> 4;
  const int lm    = lane & 15;
  const int row0  = mtile * 32;
  const int col0  = nblk * 64;

  const unsigned short* Arow0 = X + (size_t)(row0 + lm) * KK;
  const unsigned short* Arow1 = Arow0 + (size_t)16 * KK;
  v8f c[2][4] = {};

  for (int k0 = 0; k0 < KK; k0 += 32) {
    Frag a0, a1;
    a0.q[0] = *(const u32x4*)(Arow0 + k0 + half * 8);
    a0.q[1] = *(const u32x4*)(Arow0 + k0 + 16 + half * 8);
    a1.q[0] = *(const u32x4*)(Arow1 + k0 + half * 8);
    a1.q[1] = *(const u32x4*)(Arow1 + k0 + 16 + half * 8);
#pragma unroll
    for (int f = 0; f < 4; ++f) {
      const unsigned short* Wrow = W + (size_t)(col0 + f * 16 + lm) * KK + k0 + half * 16;
      Frag b;
      b.q[0] = *(const u32x4*)(Wrow - half * 8);
      b.q[1] = *(const u32x4*)(Wrow + 16 - half * 8);
      c[0][f] = WMMA_BF16(a0.bf, b.bf, c[0][f]);
      c[1][f] = WMMA_BF16(a1.bf, b.bf, c[1][f]);
    }
  }

#pragma unroll
  for (int mi = 0; mi < 2; ++mi) {
#pragma unroll
    for (int f = 0; f < 4; ++f) {
#pragma unroll
      for (int r = 0; r < 8; ++r) {
        const int row  = row0 + mi * 16 + r + half * 8;
        const int coln = col0 + f * 16 + lm;
        const float val = c[mi][f][r] + bias[coln];
        if (mode == 2) {
          outf[(size_t)row * NN + coln] = val;
        } else if (mode == 0) {
          outb[(size_t)row * NN + coln] = f32_to_bf16(val);
        } else {                                   // vT[b][h][d][s]
          const int bb = row / SS, s = row % SS;
          const int h = coln / HD, d = coln % HD;
          outb[(((size_t)(bb * HH + h)) * HD + d) * SS + s] = f32_to_bf16(val);
        }
      }
    }
  }
}

// ---------------- flash attention: one wave per (b,h,16-query tile), 64 keys/iter ----
__global__ __launch_bounds__(32) void attn_kernel(
    const unsigned short* __restrict__ q, const unsigned short* __restrict__ k,
    const unsigned short* __restrict__ vT, unsigned short* __restrict__ attn_out) {
  const int qt = blockIdx.x & 127;
  const int h  = (blockIdx.x >> 7) & 15;
  const int bb = blockIdx.x >> 11;
  const int lane = threadIdx.x;
  const int half = lane >> 4;
  const int lm   = lane & 15;

  __shared__ __align__(16) unsigned short pbuf[16 * 64];   // P tile bounce (C->A layout)

  // q A-fragments (hd=64 -> two K=32 fragments), register-resident all kernel
  Frag aq[2];
  const unsigned short* qrow = q + ((size_t)(bb * SS + qt * 16 + lm) * DD + h * HD);
#pragma unroll
  for (int kb = 0; kb < 2; ++kb) {
    aq[kb].q[0] = *(const u32x4*)(qrow + kb * 32 + half * 16);
    aq[kb].q[1] = *(const u32x4*)(qrow + kb * 32 + half * 16 + 8);
  }

  v8f acc[4] = {};
  float rmax[8], rsum[8];
#pragma unroll
  for (int r = 0; r < 8; ++r) { rmax[r] = -1e30f; rsum[r] = 0.f; }

  const float SCALE = 0.125f * 1.44269504088896f;          // 1/sqrt(64) * log2(e)

  for (int kblk = 0; kblk < SS / 64; ++kblk) {
    const int kbase = kblk * 64;

    // ---- scores for four 16-key tiles (64 keys) ----
    v8f s[4] = {};
#pragma unroll
    for (int t = 0; t < 4; ++t) {
      const unsigned short* krow =
          k + ((size_t)(bb * SS + kbase + t * 16 + lm) * DD + h * HD) + half * 16;
#pragma unroll
      for (int c = 0; c < 2; ++c) {
        Frag b;
        b.q[0] = *(const u32x4*)(krow + c * 32);
        b.q[1] = *(const u32x4*)(krow + c * 32 + 8);
        s[t] = WMMA_BF16(aq[c].bf, b.bf, s[t]);
      }
    }
#pragma unroll
    for (int t = 0; t < 4; ++t)
#pragma unroll
      for (int r = 0; r < 8; ++r) s[t][r] *= SCALE;

    // ---- online softmax stats (rows live per-VGPR within each 16-lane half) ----
    float nm[8];
#pragma unroll
    for (int r = 0; r < 8; ++r)
      nm[r] = fmaxf(fmaxf(s[0][r], s[1][r]), fmaxf(s[2][r], s[3][r]));
#pragma unroll
    for (int off = 1; off < 16; off <<= 1)
#pragma unroll
      for (int r = 0; r < 8; ++r) nm[r] = fmaxf(nm[r], __shfl_xor(nm[r], off, 32));

    float p[4][8], ts[8];
#pragma unroll
    for (int r = 0; r < 8; ++r) {
      nm[r] = fmaxf(nm[r], rmax[r]);
      const float alpha = exp2f(rmax[r] - nm[r]);
      rmax[r] = nm[r];
      ts[r] = 0.f;
#pragma unroll
      for (int t = 0; t < 4; ++t) {
        p[t][r] = exp2f(s[t][r] - nm[r]);
        ts[r] += p[t][r];
      }
      rsum[r] *= alpha;
#pragma unroll
      for (int f = 0; f < 4; ++f) acc[f][r] *= alpha;
    }
#pragma unroll
    for (int off = 1; off < 16; off <<= 1)
#pragma unroll
      for (int r = 0; r < 8; ++r) ts[r] += __shfl_xor(ts[r], off, 32);
#pragma unroll
    for (int r = 0; r < 8; ++r) rsum[r] += ts[r];

    // ---- P: C-layout -> A-layout via LDS (single wave; barriers are waitcnt-only) ----
#pragma unroll
    for (int r = 0; r < 8; ++r) {
      const int row = r + half * 8;
#pragma unroll
      for (int t = 0; t < 4; ++t)
        pbuf[row * 64 + t * 16 + lm] = f32_to_bf16(p[t][r]);
    }
    __syncthreads();
    Frag ap[2];
#pragma unroll
    for (int kb = 0; kb < 2; ++kb) {
      ap[kb].q[0] = *(const u32x4*)(&pbuf[lm * 64 + kb * 32 + half * 16]);
      ap[kb].q[1] = *(const u32x4*)(&pbuf[lm * 64 + kb * 32 + half * 16 + 8]);
    }
    __syncthreads();

    // ---- P @ V over 64 keys (vT layout -> contiguous B loads) ----
#pragma unroll
    for (int kb = 0; kb < 2; ++kb) {
#pragma unroll
      for (int f = 0; f < 4; ++f) {
        const unsigned short* vr =
            vT + (((size_t)(bb * HH + h)) * HD + f * 16 + lm) * SS + kbase + kb * 32 + half * 16;
        Frag bv;
        bv.q[0] = *(const u32x4*)(vr);
        bv.q[1] = *(const u32x4*)(vr + 8);
        acc[f] = WMMA_BF16(ap[kb].bf, bv.bf, acc[f]);
      }
    }
  }

  // ---- epilogue: normalize and store bf16 [b,s,(h,hd)] ----
#pragma unroll
  for (int r = 0; r < 8; ++r) {
    const float inv = 1.0f / rsum[r];
    const int s = qt * 16 + r + half * 8;
#pragma unroll
    for (int f = 0; f < 4; ++f) {
      const int col = h * HD + f * 16 + lm;
      attn_out[(size_t)(bb * SS + s) * DD + col] = f32_to_bf16(acc[f][r] * inv);
    }
  }
}

// ---------------- host ----------------
extern "C" void kernel_launch(void* const* d_in, const int* in_sizes, int n_in,
                              void* d_out, int out_size, void* d_ws, size_t ws_size,
                              hipStream_t stream) {
  const float* Q  = (const float*)d_in[0];
  const float* K  = (const float*)d_in[1];
  const float* V  = (const float*)d_in[2];
  const float* Wq = (const float*)d_in[3];
  const float* bq = (const float*)d_in[4];
  const float* Wk = (const float*)d_in[5];
  const float* bk = (const float*)d_in[6];
  const float* Wv = (const float*)d_in[7];
  const float* bv = (const float*)d_in[8];
  const float* Wo = (const float*)d_in[9];
  const float* bo = (const float*)d_in[10];

  // bf16 workspace (elements):
  //  Xq,Xk,Xv: 4M each | Wq,Wk,Wv,Wo: 1M each | q,k,vT: 4M each (attn overwrites q)
  const size_t ACT = (size_t)MM * KK;   // 4M
  const size_t WGT = (size_t)NN * KK;   // 1M
  unsigned short* xq = (unsigned short*)d_ws;
  unsigned short* xk = xq + ACT;
  unsigned short* xv = xk + ACT;
  unsigned short* wq = xv + ACT;
  unsigned short* wk = wq + WGT;
  unsigned short* wv = wk + WGT;
  unsigned short* wo = wv + WGT;
  unsigned short* qp = wo + WGT;
  unsigned short* kp = qp + ACT;
  unsigned short* vp = kp + ACT;

  const int actBlocks = (int)(ACT / (8 * 256));   // 2048
  const int wgtBlocks = (int)(WGT / (8 * 256));   // 512
  cvt_bf16_kernel<<<actBlocks, 256, 0, stream>>>(Q,  xq);
  cvt_bf16_kernel<<<actBlocks, 256, 0, stream>>>(K,  xk);
  cvt_bf16_kernel<<<actBlocks, 256, 0, stream>>>(V,  xv);
  cvt_bf16_kernel<<<wgtBlocks, 256, 0, stream>>>(Wq, wq);
  cvt_bf16_kernel<<<wgtBlocks, 256, 0, stream>>>(Wk, wk);
  cvt_bf16_kernel<<<wgtBlocks, 256, 0, stream>>>(Wv, wv);
  cvt_bf16_kernel<<<wgtBlocks, 256, 0, stream>>>(Wo, wo);

  // (M/32)*(N/64) = 2048 waves -> 256 blocks of 8 waves
  gemm_bf16_kernel<<<256, 256, 0, stream>>>(xq, wq, bq, qp, nullptr, 0);
  gemm_bf16_kernel<<<256, 256, 0, stream>>>(xk, wk, bk, kp, nullptr, 0);
  gemm_bf16_kernel<<<256, 256, 0, stream>>>(xv, wv, bv, vp, nullptr, 1); // vT[b][h][d][s]

  attn_kernel<<<Bb * HH * (SS / 16), 32, 0, stream>>>(qp, kp, vp, qp);   // in-place over q

  gemm_bf16_kernel<<<256, 256, 0, stream>>>(qp, wo, bo, nullptr, (float*)d_out, 2);
}
